// CAPAttentionModule_82343112998929
// MI455X (gfx1250) — compile-verified, matrix operand order checked by reading, not run on hardware
//
#include <hip/hip_runtime.h>
#include <hip/hip_bf16.h>

// ---------------------------------------------------------------------------
// CAP attention module, MI455X (gfx1250, wave32, WMMA).
// All heavy GEMMs use v_wmma_f32_16x16x32_bf16 with fp32 accumulation.
// LDS tiles are laid out K-major on both operands so every fragment is two
// contiguous 16-byte chunks -> ds_load_b128 (no scalar ds_load_u16 paths).
// ---------------------------------------------------------------------------

typedef __attribute__((ext_vector_type(16))) __bf16 v16bf;
typedef __attribute__((ext_vector_type(8)))  float  v8f;

union BF16Frag { v16bf v; uint4 q[2]; unsigned short u[16]; };

#define NB   8      // batch
#define CIN  512
#define KC   256
#define HH   96
#define WW   96
#define HWPX 9216   // 96*96
#define SDIM 110
#define SPAD 128
#define MTOT 640    // 256 query + 128 keyP + 256 valueP

__device__ __forceinline__ unsigned short f2bf(float f) {
    unsigned int u = __float_as_uint(f);
    unsigned int r = u + 0x7FFFu + ((u >> 16) & 1u);   // round-to-nearest-even
    return (unsigned short)(r >> 16);
}

// CDNA5 16-bit A/B fragment from a K-major LDS row (row = 32 bf16 of one M/N
// line). Per ISA 7.12.2 the lane's 16 elements are K = {8*hi..8*hi+7} and
// K = {16+8*hi..23+8*hi}  (hi = lane>>4) -> two aligned 16-byte chunks.
__device__ __forceinline__ v16bf load_frag(const unsigned short* row, int hi) {
    BF16Frag f;
    f.q[0] = *(const uint4*)(row + 8 * hi);
    f.q[1] = *(const uint4*)(row + 16 + 8 * hi);
    return f.v;
}

// ---------------------------------------------------------------------------
// Kernel 0: pack weights -> bf16 [640][512], fused BN affine vectors [640]
// ---------------------------------------------------------------------------
__global__ __launch_bounds__(256) void k_prep(
    const float* __restrict__ q_w, const float* __restrict__ kp_w,
    const float* __restrict__ vp_w,
    const float* __restrict__ q_g, const float* __restrict__ q_b,
    const float* __restrict__ kp_g, const float* __restrict__ kp_b,
    const float* __restrict__ vp_g, const float* __restrict__ vp_b,
    unsigned short* __restrict__ Wb, float* __restrict__ Gc, float* __restrict__ Bc)
{
    int m = blockIdx.x;
    const float *srcw, *sg, *sb; int lc;
    if (m < 256)      { srcw = q_w  + m * CIN;          sg = q_g;  sb = q_b;  lc = m; }
    else if (m < 384) { srcw = kp_w + (m - 256) * CIN;  sg = kp_g; sb = kp_b; lc = m - 256; }
    else              { srcw = vp_w + (m - 384) * CIN;  sg = vp_g; sb = vp_b; lc = m - 384; }
    for (int k = threadIdx.x; k < CIN; k += 256)
        Wb[m * CIN + k] = f2bf(srcw[k]);
    if (threadIdx.x == 0) { Gc[m] = sg[lc]; Bc[m] = sb[lc]; }
}

// ---------------------------------------------------------------------------
// Kernel 1: fused 1x1 convs as one WMMA GEMM.
//   Y[m, n] = relu(g[m] * sum_k W[m,k] * x[b,k,n] + b[m])
// Tile: 64(M) x 64(N) per WG (128 thr = 4 waves, wave n-slice of 16), BK=32.
// ---------------------------------------------------------------------------
__global__ __launch_bounds__(128) void k_gemm640(
    const float* __restrict__ x, const unsigned short* __restrict__ Wb,
    const float* __restrict__ Gc, const float* __restrict__ Bc,
    unsigned short* __restrict__ Qb, float* __restrict__ KP, float* __restrict__ VP)
{
    __shared__ alignas(16) unsigned short As[64 * 32];   // [m][k]  weights bf16
    __shared__ alignas(16) unsigned short BsT[64 * 32];  // [n][k]  x tile bf16

    const int tid  = threadIdx.x;
    const int wave = tid >> 5;
    const int lane = tid & 31;
    const int hi   = lane >> 4;
    const int lcol = lane & 15;
    const int n0   = blockIdx.x * 64;
    const int m0   = blockIdx.y * 64;
    const int b    = blockIdx.z;

    v8f acc[4];
    for (int t = 0; t < 4; ++t)
        for (int r = 0; r < 8; ++r) acc[t][r] = 0.0f;

    unsigned int* BsTu = (unsigned int*)BsT;

    for (int kk = 0; kk < CIN; kk += 32) {
        // stage A: 16 contiguous bf16 (32B) per thread, b128 in/out
        {
            int ml = tid >> 1, half = (tid & 1) * 16;
            const uint4* g = (const uint4*)(Wb + (size_t)(m0 + ml) * CIN + kk + half);
            uint4* s = (uint4*)(As + ml * 32 + half);
            s[0] = g[0]; s[1] = g[1];
        }
        // stage B transposed: pack k,k+1 bf16 into one dword per (n, kpair)
        for (int i = 0; i < 8; ++i) {
            int idx = tid + i * 128;
            int nl = idx & 63, kp = idx >> 6;
            const float* px = x + ((size_t)(b * CIN + kk + 2 * kp)) * HWPX + n0 + nl;
            unsigned int lo = f2bf(px[0]);
            unsigned int hb = f2bf(px[HWPX]);
            BsTu[nl * 16 + kp] = lo | (hb << 16);
        }
        if (kk + 32 < CIN) {   // prefetch next K-step tiles
            __builtin_prefetch(x + ((size_t)(b * CIN + kk + 32 + (tid >> 2))) * HWPX + n0 + ((tid & 3) << 4), 0, 1);
            __builtin_prefetch(Wb + (size_t)(m0 + (tid >> 1)) * CIN + kk + 32, 0, 1);
        }
        __syncthreads();

        v16bf bfv = load_frag(BsT + (16 * wave + lcol) * 32, hi);
        for (int t = 0; t < 4; ++t) {
            v16bf afv = load_frag(As + (16 * t + lcol) * 32, hi);
            acc[t] = __builtin_amdgcn_wmma_f32_16x16x32_bf16(
                false, afv, false, bfv, (short)0, acc[t], false, false);
        }
        __syncthreads();
    }

    // epilogue: BN affine + ReLU, route to query(bf16) / keyP(f32) / valueP(f32)
    const int n = n0 + 16 * wave + lcol;
    for (int t = 0; t < 4; ++t) {
        for (int r = 0; r < 8; ++r) {
            int m = m0 + 16 * t + r + 8 * hi;
            float y = fmaxf(acc[t][r] * Gc[m] + Bc[m], 0.0f);
            if (m < 256)
                Qb[((size_t)(b * KC + m)) * HWPX + n] = f2bf(y);
            else if (m < 384)
                KP[((size_t)(b * 128 + (m - 256))) * HWPX + n] = y;
            else
                VP[((size_t)(b * 256 + (m - 384))) * HWPX + n] = y;
        }
    }
}

// ---------------------------------------------------------------------------
// Kernel 2: depthwise 3x3 + BN + ReLU (ghost branch) fused with PSP pooling
// (sizes 1,3,6,8 -> 110 bins) for BOTH the primary and ghost channels.
// Hierarchical 4x4-block partial sums in LDS. Output bf16, padded to S=128.
// ---------------------------------------------------------------------------
__global__ __launch_bounds__(256) void k_dwpool(
    const float* __restrict__ KP, const float* __restrict__ VP,
    const float* __restrict__ kc_w, const float* __restrict__ kc_g, const float* __restrict__ kc_b,
    const float* __restrict__ vc_w, const float* __restrict__ vc_g, const float* __restrict__ vc_b,
    unsigned short* __restrict__ Kb, unsigned short* __restrict__ Vb)
{
    __shared__ float plane[HWPX];       // 36 KB
    __shared__ float partP[576];        // 24x24 grid of 4x4-block sums (primary)
    __shared__ float partD[576];        // (ghost/dw)

    const int pc  = blockIdx.x;         // 0..127 key primary, 128..383 value primary
    const int b   = blockIdx.y;
    const int tid = threadIdx.x;

    const float* src; const float* w9; float gg, bb;
    unsigned short *outP, *outD;
    if (pc < 128) {
        src = KP + ((size_t)(b * 128 + pc)) * HWPX;
        w9 = kc_w + pc * 9; gg = kc_g[pc]; bb = kc_b[pc];
        outP = Kb + ((size_t)(b * KC + pc)) * SPAD;
        outD = Kb + ((size_t)(b * KC + 128 + pc)) * SPAD;
    } else {
        int c = pc - 128;
        src = VP + ((size_t)(b * 256 + c)) * HWPX;
        w9 = vc_w + c * 9; gg = vc_g[c]; bb = vc_b[c];
        outP = Vb + ((size_t)(b * CIN + c)) * SPAD;
        outD = Vb + ((size_t)(b * CIN + 256 + c)) * SPAD;
    }

    for (int i = 0; i < HWPX / 256; ++i) plane[tid + i * 256] = src[tid + i * 256];
    float w00 = w9[0], w01 = w9[1], w02 = w9[2];
    float w10 = w9[3], w11 = w9[4], w12 = w9[5];
    float w20 = w9[6], w21 = w9[7], w22 = w9[8];
    __syncthreads();

    for (int pb = tid; pb < 576; pb += 256) {
        int by = pb / 24, bx = pb % 24;
        float sp = 0.0f, sd = 0.0f;
        for (int dy = 0; dy < 4; ++dy) {
            int h = by * 4 + dy;
            for (int dx = 0; dx < 4; ++dx) {
                int w = bx * 4 + dx;
                sp += plane[h * WW + w];
                float a = 0.0f;
                #pragma unroll
                for (int iy = -1; iy <= 1; ++iy) {
                    int hh = h + iy;
                    if (hh < 0 || hh >= HH) continue;
                    #pragma unroll
                    for (int ix = -1; ix <= 1; ++ix) {
                        int ww = w + ix;
                        if (ww < 0 || ww >= WW) continue;
                        float wt = (iy < 0) ? ((ix < 0) ? w00 : (ix == 0) ? w01 : w02)
                                 : (iy == 0) ? ((ix < 0) ? w10 : (ix == 0) ? w11 : w12)
                                              : ((ix < 0) ? w20 : (ix == 0) ? w21 : w22);
                        a += wt * plane[hh * WW + ww];
                    }
                }
                sd += fmaxf(gg * a + bb, 0.0f);
            }
        }
        partP[pb] = sp; partD[pb] = sd;
    }
    __syncthreads();

    if (tid < 220) {
        int arr = tid / SDIM;        // 0 primary, 1 ghost
        int bin = tid % SDIM;
        int s, idx;
        if (bin == 0)      { s = 1; idx = 0; }
        else if (bin < 10) { s = 3; idx = bin - 1; }
        else if (bin < 46) { s = 6; idx = bin - 10; }
        else               { s = 8; idx = bin - 46; }
        int gy = idx / s, gx = idx % s;
        int bs = HH / s;             // 96,32,16,12
        int nb = bs / 4;             // 24,8,4,3
        const float* part = arr ? partD : partP;
        float sum = 0.0f;
        for (int py = 0; py < nb; ++py)
            for (int px = 0; px < nb; ++px)
                sum += part[(gy * nb + py) * 24 + gx * nb + px];
        (arr ? outD : outP)[bin] = f2bf(sum / (float)(bs * bs));
    } else {
        int k = tid - 220;           // 36 pad slots: 2 arrays x 18
        if (k < 36) {
            int arr = k / 18;
            (arr ? outD : outP)[SDIM + (k % 18)] = 0;
        }
    }
}

// ---------------------------------------------------------------------------
// Kernel 3: sim = (Q^T K) / 16, softmax over S, store P^T bf16 [S=128][pixels].
// WG = 128 thr (4 waves); WG tile = 64 pixels x 128 S; wave = 64(M) x 32(N).
// ---------------------------------------------------------------------------
__global__ __launch_bounds__(128) void k_attn_sim(
    const unsigned short* __restrict__ Qb, const unsigned short* __restrict__ Kb,
    unsigned short* __restrict__ PT)
{
    __shared__ alignas(16) unsigned short Qs[64 * 32];     // [m][k]
    __shared__ alignas(16) unsigned short KsT[SPAD * 32];  // [s][k]
    __shared__ float simS[64 * SPAD];                      // 32 KB

    const int tid  = threadIdx.x;
    const int wave = tid >> 5;
    const int lane = tid & 31;
    const int hi   = lane >> 4;
    const int lcol = lane & 15;
    const int m0   = blockIdx.x * 64;         // pixel block
    const int b    = blockIdx.y;

    v8f acc[4][2];
    for (int mt = 0; mt < 4; ++mt)
        for (int nt = 0; nt < 2; ++nt)
            for (int r = 0; r < 8; ++r) acc[mt][nt][r] = 0.0f;

    unsigned int* Qsu = (unsigned int*)Qs;
    unsigned int* Ksu = (unsigned int*)KsT;

    for (int kk = 0; kk < KC; kk += 32) {
        for (int i = 0; i < 8; ++i) {         // Q[k][pixel] -> Qs[m][k], 2k/dword
            int idx = tid + i * 128;
            int ml = idx & 63, kp = idx >> 6;
            const unsigned short* q = Qb + ((size_t)(b * KC + kk + 2 * kp)) * HWPX + m0 + ml;
            Qsu[ml * 16 + kp] = (unsigned int)q[0] | ((unsigned int)q[HWPX] << 16);
        }
        for (int i = 0; i < 16; ++i) {        // K[k][s] -> KsT[s][k], 2k/dword
            int idx = tid + i * 128;
            int sl = idx & 127, kp = idx >> 7;
            const unsigned short* kq = Kb + ((size_t)(b * KC + kk + 2 * kp)) * SPAD + sl;
            Ksu[sl * 16 + kp] = (unsigned int)kq[0] | ((unsigned int)kq[SPAD] << 16);
        }
        if (kk + 32 < KC)
            __builtin_prefetch(Qb + ((size_t)(b * KC + kk + 32 + (tid >> 2))) * HWPX + m0, 0, 1);
        __syncthreads();

        v16bf bfv[2];
        for (int nt = 0; nt < 2; ++nt)
            bfv[nt] = load_frag(KsT + (32 * wave + 16 * nt + lcol) * 32, hi);
        for (int mt = 0; mt < 4; ++mt) {
            v16bf afv = load_frag(Qs + (16 * mt + lcol) * 32, hi);
            for (int nt = 0; nt < 2; ++nt)
                acc[mt][nt] = __builtin_amdgcn_wmma_f32_16x16x32_bf16(
                    false, afv, false, bfv[nt], (short)0, acc[mt][nt], false, false);
        }
        __syncthreads();
    }

    // spill scaled sim to LDS
    for (int mt = 0; mt < 4; ++mt)
        for (int nt = 0; nt < 2; ++nt)
            for (int r = 0; r < 8; ++r) {
                int ml = 16 * mt + r + 8 * hi;
                int nl = 32 * wave + 16 * nt + lcol;
                simS[ml * SPAD + nl] = acc[mt][nt][r] * 0.0625f;   // Kc^-0.5
            }
    __syncthreads();

    // softmax over s < 110 per pixel row; pad region -> 0 probability
    if (tid < 64) {
        float* row = simS + tid * SPAD;
        float mx = -3.4e38f;
        for (int s = 0; s < SDIM; ++s) mx = fmaxf(mx, row[s]);
        float sum = 0.0f;
        for (int s = 0; s < SDIM; ++s) { float e = __expf(row[s] - mx); row[s] = e; sum += e; }
        float inv = 1.0f / sum;
        for (int s = 0; s < SDIM; ++s) row[s] *= inv;
        for (int s = SDIM; s < SPAD; ++s) row[s] = 0.0f;
    }
    __syncthreads();

    // store transposed probabilities PT[b][s][pixel], 2 pixels per dword
    for (int i = 0; i < 32; ++i) {
        int idx = tid + 128 * i;
        int jp = idx & 31, s = idx >> 5;
        unsigned int lo = f2bf(simS[(2 * jp) * SPAD + s]);
        unsigned int hb = f2bf(simS[(2 * jp + 1) * SPAD + s]);
        *(unsigned int*)(PT + ((size_t)(b * SPAD + s)) * HWPX + m0 + 2 * jp) = lo | (hb << 16);
    }
}

// ---------------------------------------------------------------------------
// Kernel 4: context GEMM ctx[c,n] = sum_s V[c,s] * P[n,s]  (K = 128 padded),
// plus residual: out = x + ctx.
// ---------------------------------------------------------------------------
__global__ __launch_bounds__(128) void k_context(
    const float* __restrict__ x, const unsigned short* __restrict__ Vb,
    const unsigned short* __restrict__ PT, float* __restrict__ out)
{
    __shared__ alignas(16) unsigned short As[64 * 32];   // [c][s] V tile
    __shared__ alignas(16) unsigned short BsT[64 * 32];  // [n][s] P tile

    const int tid  = threadIdx.x;
    const int wave = tid >> 5;
    const int lane = tid & 31;
    const int hi   = lane >> 4;
    const int lcol = lane & 15;
    const int n0   = blockIdx.x * 64;
    const int c0   = blockIdx.y * 64;
    const int b    = blockIdx.z;

    v8f acc[4];
    for (int t = 0; t < 4; ++t)
        for (int r = 0; r < 8; ++r) acc[t][r] = 0.0f;

    unsigned int* BsTu = (unsigned int*)BsT;

    for (int kk = 0; kk < SPAD; kk += 32) {
        {
            int ml = tid >> 1, half = (tid & 1) * 16;
            const uint4* g = (const uint4*)(Vb + (size_t)(b * CIN + c0 + ml) * SPAD + kk + half);
            uint4* s = (uint4*)(As + ml * 32 + half);
            s[0] = g[0]; s[1] = g[1];
        }
        for (int i = 0; i < 8; ++i) {
            int idx = tid + i * 128;
            int nl = idx & 63, kp = idx >> 6;
            const unsigned short* pp = PT + ((size_t)(b * SPAD + kk + 2 * kp)) * HWPX + n0 + nl;
            BsTu[nl * 16 + kp] = (unsigned int)pp[0] | ((unsigned int)pp[HWPX] << 16);
        }
        if (kk + 32 < SPAD)
            __builtin_prefetch(PT + ((size_t)(b * SPAD + kk + 32 + (tid >> 2))) * HWPX + n0, 0, 1);
        __syncthreads();

        v16bf bfv = load_frag(BsT + (16 * wave + lcol) * 32, hi);
        for (int t = 0; t < 4; ++t) {
            v16bf afv = load_frag(As + (16 * t + lcol) * 32, hi);
            acc[t] = __builtin_amdgcn_wmma_f32_16x16x32_bf16(
                false, afv, false, bfv, (short)0, acc[t], false, false);
        }
        __syncthreads();
    }

    const int n = n0 + 16 * wave + lcol;
    for (int t = 0; t < 4; ++t)
        for (int r = 0; r < 8; ++r) {
            int c = c0 + 16 * t + r + 8 * hi;
            size_t gi = ((size_t)(b * CIN + c)) * HWPX + n;
            out[gi] = x[gi] + acc[t][r];
        }
}

// ---------------------------------------------------------------------------
// Launch
// ---------------------------------------------------------------------------
extern "C" void kernel_launch(void* const* d_in, const int* in_sizes, int n_in,
                              void* d_out, int out_size, void* d_ws, size_t ws_size,
                              hipStream_t stream)
{
    const float* x    = (const float*)d_in[0];
    const float* q_w  = (const float*)d_in[1];
    const float* q_g  = (const float*)d_in[2];
    const float* q_b  = (const float*)d_in[3];
    const float* kp_w = (const float*)d_in[4];
    const float* kp_g = (const float*)d_in[5];
    const float* kp_b = (const float*)d_in[6];
    const float* kc_w = (const float*)d_in[7];
    const float* kc_g = (const float*)d_in[8];
    const float* kc_b = (const float*)d_in[9];
    const float* vp_w = (const float*)d_in[10];
    const float* vp_g = (const float*)d_in[11];
    const float* vp_b = (const float*)d_in[12];
    const float* vc_w = (const float*)d_in[13];
    const float* vc_g = (const float*)d_in[14];
    const float* vc_b = (const float*)d_in[15];
    float* out = (float*)d_out;

    char* ws = (char*)d_ws;
    size_t off = 0;
    auto take = [&](size_t bytes) { char* p = ws + off; off += (bytes + 255) & ~(size_t)255; return p; };
    unsigned short* Wb = (unsigned short*)take((size_t)MTOT * CIN * 2);        // 640x512 bf16
    float*          Gc = (float*)take((size_t)MTOT * 4);
    float*          Bc = (float*)take((size_t)MTOT * 4);
    unsigned short* Qb = (unsigned short*)take((size_t)NB * KC * HWPX * 2);    // query bf16
    float*          KP = (float*)take((size_t)NB * 128 * HWPX * 4);            // key primary f32
    float*          VP = (float*)take((size_t)NB * 256 * HWPX * 4);            // value primary f32
    unsigned short* Kb = (unsigned short*)take((size_t)NB * KC * SPAD * 2);    // pooled key bf16
    unsigned short* Vb = (unsigned short*)take((size_t)NB * CIN * SPAD * 2);   // pooled value bf16
    unsigned short* PT = (unsigned short*)take((size_t)NB * SPAD * HWPX * 2);  // probs^T bf16

    k_prep<<<dim3(MTOT), dim3(256), 0, stream>>>(q_w, kp_w, vp_w, q_g, q_b,
                                                 kp_g, kp_b, vp_g, vp_b, Wb, Gc, Bc);

    k_gemm640<<<dim3(HWPX / 64, MTOT / 64, NB), dim3(128), 0, stream>>>(
        x, Wb, Gc, Bc, Qb, KP, VP);

    k_dwpool<<<dim3(384, NB), dim3(256), 0, stream>>>(
        KP, VP, kc_w, kc_g, kc_b, vc_w, vc_g, vc_b, Kb, Vb);

    k_attn_sim<<<dim3(HWPX / 64, NB), dim3(128), 0, stream>>>(Qb, Kb, PT);

    k_context<<<dim3(HWPX / 64, CIN / 64, NB), dim3(128), 0, stream>>>(x, Vb, PT, out);
}
